// GNNEncoder_82403242541376
// MI455X (gfx1250) — matrix operand order repaired and checked by reading, not compiled, not run on hardware
//
#include <hip/hip_runtime.h>
#include <hip/hip_bf16.h>

typedef __attribute__((ext_vector_type(16))) _Float16 v16h;
typedef __attribute__((ext_vector_type(8)))  float    v8f;
typedef __attribute__((ext_vector_type(2)))  float    v2f;

#define BATCH 128
#define NPTS  512
#define KNN   16
#define ESTRIDE 264   // LDS row stride (f16) for E and H tiles, even, >= 256
#define DSTRIDE 520   // LDS row stride (f32) for distance rows, 520%64==8

// ---------------------------------------------------------------------------
// Top-16 register insertion: guarded, fully unrolled swap-down chain.
// ---------------------------------------------------------------------------
__device__ __forceinline__ void topk_insert(float (&bd)[KNN], int (&bj)[KNN],
                                            float dv, int jv) {
    if (dv < bd[KNN - 1]) {
        float cd = dv; int cj = jv;
#pragma unroll
        for (int r = 0; r < KNN; ++r) {
            if (cd < bd[r]) {
                float td = bd[r]; bd[r] = cd; cd = td;
                int   tj = bj[r]; bj[r] = cj; cj = tj;
            }
        }
    }
}

// ---------------------------------------------------------------------------
// Squared norms, one thread per node.
// ---------------------------------------------------------------------------
__global__ void sqnorm_kernel(const float* __restrict__ feat,
                              float* __restrict__ sqn, int d) {
    int gi = blockIdx.x * blockDim.x + threadIdx.x;
    const float* x = feat + (size_t)gi * d;
    float s = 0.f;
    for (int c = 0; c < d; ++c) s = fmaf(x[c], x[c], s);
    sqn[gi] = s;
}

// ---------------------------------------------------------------------------
// kNN via f32 WMMA Gram tiles. One block (64 thr = 2 waves) per (b, 16-row
// tile). A fragments for the 16 query rows are register-resident (templated
// on D); each wave sweeps half the 16-column candidate tiles, K in chunks of
// 4 via V_WMMA_F32_16X16X4_F32. dist = |xi|^2 + |xj|^2 - 2*Gram -> LDS, then
// 4 threads per row do a strided top-16 scan and threads 0-15 merge.
// ---------------------------------------------------------------------------
template<int D>
__global__ void knn_wmma_kernel(const float* __restrict__ feat,
                                const float* __restrict__ sqn,
                                int* __restrict__ idx, int N) {
    constexpr int D4 = (D + 3) / 4;
    __shared__ float dist_s[16][DSTRIDE];
    __shared__ float candd[16][4][KNN];
    __shared__ int   candj[16][4][KNN];

    const int t    = threadIdx.x;          // 0..63
    const int lane = t & 31;
    const int wv   = t >> 5;
    const int tilesPerB = N / 16;
    const int b    = blockIdx.x / tilesPerB;
    const int i0   = (blockIdx.x % tilesPerB) * 16;

    const int n  = lane & 15;              // row for A, column for B
    const int hi = lane >> 4;
    const float* xb  = feat + (size_t)b * N * D;
    const float* sqb = sqn + (size_t)b * N;

    // A fragments: 16x4 f32 tiles. lanes 0-15: M=lane, K=4kc+{0,1};
    // lanes 16-31: M=lane-16, K=4kc+{2,3}.
    v2f afr[D4];
#pragma unroll
    for (int kc = 0; kc < D4; ++kc) {
        int k0 = kc * 4 + hi * 2;
        float a0 = (k0     < D) ? xb[(size_t)(i0 + n) * D + k0]     : 0.f;
        float a1 = (k0 + 1 < D) ? xb[(size_t)(i0 + n) * D + k0 + 1] : 0.f;
        v2f av; av.x = a0; av.y = a1;
        afr[kc] = av;
    }
    float sqi[8];
#pragma unroll
    for (int r = 0; r < 8; ++r) sqi[r] = sqb[i0 + r + hi * 8];

    for (int jt = wv; jt < N / 16; jt += 2) {
        const int j = jt * 16 + n;
        v8f acc = {};
#pragma unroll
        for (int kc = 0; kc < D4; ++kc) {
            int k0 = kc * 4 + hi * 2;
            float b0 = (k0     < D) ? xb[(size_t)j * D + k0]     : 0.f;
            float b1 = (k0 + 1 < D) ? xb[(size_t)j * D + k0 + 1] : 0.f;
            v2f bfr; bfr.x = b0; bfr.y = b1;
            acc = __builtin_amdgcn_wmma_f32_16x16x4_f32(false, afr[kc], false, bfr,
                                                        (short)0, acc, false, false);
        }
        float sqj = sqb[j];
#pragma unroll
        for (int r = 0; r < 8; ++r) {
            dist_s[r + hi * 8][jt * 16 + n] = sqi[r] + sqj - 2.f * acc[r];
        }
    }
    __syncthreads();

    // Phase 2a: 4 threads per row, strided scan (conflict-friendly).
    {
        const int row = t >> 2, seg = t & 3;
        float bd[KNN]; int bj[KNN];
#pragma unroll
        for (int r = 0; r < KNN; ++r) { bd[r] = 3.0e38f; bj[r] = 0; }
        for (int j = seg; j < N; j += 4)
            topk_insert(bd, bj, dist_s[row][j], j);
#pragma unroll
        for (int r = 0; r < KNN; ++r) { candd[row][seg][r] = bd[r]; candj[row][seg][r] = bj[r]; }
    }
    __syncthreads();

    // Phase 2b: merge the 4 partial lists per row.
    if (t < 16) {
        float bd[KNN]; int bj[KNN];
#pragma unroll
        for (int r = 0; r < KNN; ++r) { bd[r] = 3.0e38f; bj[r] = 0; }
        for (int s = 0; s < 4; ++s)
#pragma unroll
            for (int r = 0; r < KNN; ++r)
                topk_insert(bd, bj, candd[t][s][r], candj[t][s][r]);
#pragma unroll
        for (int r = 0; r < KNN; ++r)
            idx[(size_t)(b * N + i0 + t) * KNN + r] = bj[r];
    }
}

// ---------------------------------------------------------------------------
// Pack a row-major f32 weight matrix W[Krows x h] into per-fragment f16 order:
// fragment (kt, nt) holds 32 lanes x 16 f16 contiguous. Lane's 16 values are
// k = kt*32 + (lane>=16 ? 16 : 0) + q, column n = nt*16 + (lane&15).
// ---------------------------------------------------------------------------
__global__ void pack_weights(const float* __restrict__ W, _Float16* __restrict__ Wp,
                             int Krows, int h, int kt, int nt) {
    int tid   = blockIdx.x * blockDim.x + threadIdx.x;
    int total = kt * nt * 32;
    if (tid >= total) return;
    int lane = tid & 31;
    int tile = tid >> 5;
    int ntI  = tile % nt;
    int ktI  = tile / nt;
    int n    = lane & 15;
    int kb   = (lane >= 16) ? 16 : 0;
    _Float16* dst = Wp + (size_t)tid * 16;
#pragma unroll
    for (int q = 0; q < 16; ++q) {
        int k = ktI * 32 + kb + q;
        float v = (k < Krows) ? W[(size_t)k * h + ntI * 16 + n] : 0.f;
        dst[q] = (_Float16)v;
    }
}

// ---------------------------------------------------------------------------
// A-fragment loader: 16x32 f16 A tile from row-major LDS (row stride in f16).
// ---------------------------------------------------------------------------
__device__ __forceinline__ v16h load_a_frag(const _Float16* base, int strideH,
                                            int kt, int lane) {
    const int m  = lane & 15;
    const int hi = (lane >> 4) & 1;
    const unsigned* row = (const unsigned*)(base + m * strideH);
    union { v16h v; unsigned u[8]; } a;
#pragma unroll
    for (int r = 0; r < 8; ++r) {
        int uidx = kt * 16 + (hi ? 8 : 0) + r;
        a.u[r] = row[uidx];
    }
    return a.v;
}

// ---------------------------------------------------------------------------
// EdgeConv MLP: one wave per node, 2 nodes per 64-thread block.
// N-dimension blocked by 4 tiles: one A-fragment LDS load feeds 4 WMMAs.
// ---------------------------------------------------------------------------
__global__ void edge_mlp_kernel(const float* __restrict__ feat,
                                const int* __restrict__ idx,
                                const _Float16* __restrict__ W1p,
                                const float* __restrict__ b1,
                                const _Float16* __restrict__ W2p,
                                const float* __restrict__ b2,
                                float* __restrict__ out,
                                int N, int d_in, int h,
                                int kt1, int kt2, int ntiles) {
    __shared__ __align__(16) _Float16 Es[2][16][ESTRIDE];
    __shared__ __align__(16) _Float16 Hs[2][16][ESTRIDE];

    const int lane = threadIdx.x & 31;
    const int wv   = threadIdx.x >> 5;
    const int node = blockIdx.x * 2 + wv;              // exact grid: no guard
    const int b    = node / N;
    const int i    = node % N;

    const int K1 = kt1 * 32;
    const float* fi  = feat + (size_t)(b * N + i) * d_in;
    const int*   nbr = idx + (size_t)node * KNN;

    // Build edge matrix E = [x_i | x_j - x_i] (zero-padded K columns).
    for (int e = lane; e < 16 * K1; e += 32) {
        int r = e / K1;
        int c = e % K1;
        float v = 0.f;
        if (c < d_in) {
            v = fi[c];
        } else if (c < 2 * d_in) {
            int cc = c - d_in;
            int j  = nbr[r];
            v = feat[(size_t)(b * N + j) * d_in + cc] - fi[cc];
        }
        Es[wv][r][c] = (_Float16)v;
    }
    __syncthreads();

    const int n  = lane & 15;
    const int hi = lane >> 4;
    const int groups = ntiles >> 2;

    // GEMM1: H1 = relu(E @ W1 + b1)
    for (int g = 0; g < groups; ++g) {
        v8f acc[4] = {{}, {}, {}, {}};
        for (int t = 0; t < kt1; ++t) {
            v16h a = load_a_frag(&Es[wv][0][0], ESTRIDE, t, lane);
            const _Float16* wb = W1p + ((size_t)(t * ntiles + g * 4) * 32 + lane) * 16;
#pragma unroll
            for (int q = 0; q < 4; ++q) {
                v16h bb = *(const v16h*)(wb + (size_t)q * 32 * 16);
                acc[q] = __builtin_amdgcn_wmma_f32_16x16x32_f16(false, a, false, bb,
                                                                (short)0, acc[q], false, false);
            }
        }
#pragma unroll
        for (int q = 0; q < 4; ++q) {
            int nt = g * 4 + q;
            float bias = b1[nt * 16 + n];
#pragma unroll
            for (int r = 0; r < 8; ++r) {
                float v = fmaxf(acc[q][r] + bias, 0.f);
                Hs[wv][r + (hi ? 8 : 0)][nt * 16 + n] = (_Float16)v;
            }
        }
    }
    __syncthreads();

    // GEMM2: H2 = relu(H1 @ W2 + b2), then mean over the 16 edge rows.
    for (int g = 0; g < groups; ++g) {
        v8f acc[4] = {{}, {}, {}, {}};
        for (int t = 0; t < kt2; ++t) {
            v16h a = load_a_frag(&Hs[wv][0][0], ESTRIDE, t, lane);
            const _Float16* wb = W2p + ((size_t)(t * ntiles + g * 4) * 32 + lane) * 16;
#pragma unroll
            for (int q = 0; q < 4; ++q) {
                v16h bb = *(const v16h*)(wb + (size_t)q * 32 * 16);
                acc[q] = __builtin_amdgcn_wmma_f32_16x16x32_f16(false, a, false, bb,
                                                                (short)0, acc[q], false, false);
            }
        }
#pragma unroll
        for (int q = 0; q < 4; ++q) {
            int nt = g * 4 + q;
            float bias = b2[nt * 16 + n];
            float s = 0.f;
#pragma unroll
            for (int r = 0; r < 8; ++r) s += fmaxf(acc[q][r] + bias, 0.f);
            s += __shfl_xor(s, 16);                    // fold M 0-7 with M 8-15
            if (hi == 0)
                out[(size_t)(b * N + i) * h + nt * 16 + n] = s * (1.0f / 16.0f);
        }
    }
}

// ---------------------------------------------------------------------------
// Head: mean over N, then 256 -> 128 (relu) -> 8. One block per batch element.
// ---------------------------------------------------------------------------
__global__ void head_kernel(const float* __restrict__ feat,
                            const float* __restrict__ Wf1, const float* __restrict__ bf1,
                            const float* __restrict__ Wf2, const float* __restrict__ bf2,
                            float* __restrict__ out, int N) {
    __shared__ float g[256];
    __shared__ float hbuf[128];
    const int b = blockIdx.x;
    const int t = threadIdx.x;                        // 256 threads

    float s = 0.f;
    for (int nn = 0; nn < N; ++nn) s += feat[(size_t)(b * N + nn) * 256 + t];
    g[t] = s / (float)N;
    __syncthreads();

    if (t < 128) {
        float a = bf1[t];
        for (int c = 0; c < 256; ++c) a = fmaf(g[c], Wf1[c * 128 + t], a);
        hbuf[t] = fmaxf(a, 0.f);
    }
    __syncthreads();

    if (t < 8) {
        float a = bf2[t];
        for (int c = 0; c < 128; ++c) a = fmaf(hbuf[c], Wf2[c * 8 + t], a);
        out[b * 8 + t] = a;
    }
}

// ---------------------------------------------------------------------------
// Host orchestration.
// Input order: x, (W1_l, b1_l, W2_l, b2_l) for l=0..2, Wf1, bf1, Wf2, bf2, k
// ---------------------------------------------------------------------------
extern "C" void kernel_launch(void* const* d_in, const int* in_sizes, int n_in,
                              void* d_out, int out_size, void* d_ws, size_t ws_size,
                              hipStream_t stream) {
    (void)in_sizes; (void)n_in; (void)out_size; (void)ws_size;
    const int B = BATCH, N = NPTS;

    const float* x = (const float*)d_in[0];
    const float* W1[3] = { (const float*)d_in[1], (const float*)d_in[5], (const float*)d_in[9]  };
    const float* B1[3] = { (const float*)d_in[2], (const float*)d_in[6], (const float*)d_in[10] };
    const float* W2[3] = { (const float*)d_in[3], (const float*)d_in[7], (const float*)d_in[11] };
    const float* B2[3] = { (const float*)d_in[4], (const float*)d_in[8], (const float*)d_in[12] };
    const float* Wf1 = (const float*)d_in[13];
    const float* bf1 = (const float*)d_in[14];
    const float* Wf2 = (const float*)d_in[15];
    const float* bf2 = (const float*)d_in[16];

    // Workspace carve (aligned to 256B).
    char* w = (char*)d_ws;
    auto carve = [&](size_t bytes) {
        char* p = w;
        w += (bytes + 255) & ~(size_t)255;
        return (void*)p;
    };
    float*     featA  = (float*)carve((size_t)B * N * 256 * sizeof(float));
    float*     featB  = (float*)carve((size_t)B * N * 256 * sizeof(float));
    int*       idxbuf = (int*)  carve((size_t)B * N * KNN * sizeof(int));
    float*     sqnbuf = (float*)carve((size_t)B * N * sizeof(float));
    _Float16*  wpack[6];
    for (int m = 0; m < 6; ++m)
        wpack[m] = (_Float16*)carve((size_t)8 * 16 * 32 * 16 * sizeof(_Float16)); // max kt*nt*32*16

    const int d_ins[3] = { 6, 64, 128 };
    const int hs[3]    = { 64, 128, 256 };

    const float* fin = x;
    float* bufs[2] = { featA, featB };

    for (int l = 0; l < 3; ++l) {
        const int d_in_l = d_ins[l];
        const int h      = hs[l];
        const int kt1    = (2 * d_in_l + 31) / 32;
        const int kt2    = h / 32;
        const int nt     = h / 16;
        float* fout = bufs[l & 1];

        sqnorm_kernel<<<(B * N) / 256, 256, 0, stream>>>(fin, sqnbuf, d_in_l);
        if (l == 0)
            knn_wmma_kernel<6>  <<<B * (N / 16), 64, 0, stream>>>(fin, sqnbuf, idxbuf, N);
        else if (l == 1)
            knn_wmma_kernel<64> <<<B * (N / 16), 64, 0, stream>>>(fin, sqnbuf, idxbuf, N);
        else
            knn_wmma_kernel<128><<<B * (N / 16), 64, 0, stream>>>(fin, sqnbuf, idxbuf, N);

        int tot1 = kt1 * nt * 32;
        pack_weights<<<(tot1 + 127) / 128, 128, 0, stream>>>(W1[l], wpack[2 * l],
                                                             2 * d_in_l, h, kt1, nt);
        int tot2 = kt2 * nt * 32;
        pack_weights<<<(tot2 + 127) / 128, 128, 0, stream>>>(W2[l], wpack[2 * l + 1],
                                                             h, h, kt2, nt);

        edge_mlp_kernel<<<(B * N) / 2, 64, 0, stream>>>(fin, idxbuf,
                                                        wpack[2 * l], B1[l],
                                                        wpack[2 * l + 1], B2[l],
                                                        fout, N, d_in_l, h,
                                                        kt1, kt2, nt);
        fin = fout;
    }

    head_kernel<<<B, 256, 0, stream>>>(fin, Wf1, bf1, Wf2, bf2, (float*)d_out, N);
}
